// GaussianRNNPolicy_24464133718895
// MI455X (gfx1250) — matrix operand order repaired and checked by reading, not compiled, not run on hardware
//
#include <hip/hip_runtime.h>

// ---------------------------------------------------------------------------
// GaussianRNNPolicy for MI455X (gfx1250, wave32, WMMA)
//   B=256, T=1024, D=64, H=128, Z=12, DIN=129, 3H=384
// Phase 1: gi[t*256+b][g] = x(b,t) @ W_ih^T + b_ih   (f16 WMMA, f32 accum)
// Phase 2: sequential GRU scan, W_hh fragments resident in VGPRs
// ---------------------------------------------------------------------------

typedef __attribute__((ext_vector_type(16))) _Float16 v16h;
typedef __attribute__((ext_vector_type(8)))  _Float16 v8h;
typedef __attribute__((ext_vector_type(8)))  float    v8f;

#define WMMA_F16(a, b, c) \
  __builtin_amdgcn_wmma_f32_16x16x32_f16(false, (a), false, (b), (short)0, (c), false, false)

static __device__ __forceinline__ v16h cat8(v8h a, v8h b) {
  return __builtin_shufflevector(a, b, 0,1,2,3,4,5,6,7,8,9,10,11,12,13,14,15);
}

// ---------------------------------------------------------------------------
// Phase 1: gi GEMM.  grid = 256 WGs x 256 thr (8 waves). Each wave: 8 strips
// of 16 rows (row = t*256 + b, t-major so the scan streams linearly).
// K=128 = [values*masks(64) | masks(64)]; time column + bias seed the accum.
// ---------------------------------------------------------------------------
__global__ __launch_bounds__(256) void gi_gemm_kernel(
    const float* __restrict__ times, const float* __restrict__ values,
    const float* __restrict__ masks, const float* __restrict__ W_ih,
    const float* __restrict__ b_ih, float* __restrict__ gi)
{
  extern __shared__ __align__(16) char smem[];
  _Float16* Wi    = (_Float16*)smem;                  // [384][128] f16, 96 KB
  float*    wlast = (float*)(smem + 384 * 128 * 2);   // W_ih[:,128]
  float*    bias  = wlast + 384;                      // b_ih

  for (int i = threadIdx.x; i < 384 * 128; i += 256) {
    int g = i >> 7, k = i & 127;
    Wi[i] = (_Float16)W_ih[g * 129 + k];
  }
  for (int i = threadIdx.x; i < 384; i += 256) {
    wlast[i] = W_ih[i * 129 + 128];
    bias[i]  = b_ih[i];
  }
  __syncthreads();

  const int lane = threadIdx.x & 31;
  const int wave = threadIdx.x >> 5;
  const int hi   = lane >> 4;      // half-wave select
  const int mlo  = lane & 15;

  for (int s = 0; s < 8; ++s) {
    int strip = (blockIdx.x * 8 + wave) * 8 + s;   // 0..16383
    int t     = strip >> 4;
    int btile = strip & 15;
    int b     = btile * 16 + mlo;
    size_t vbase = ((size_t)b * 1024 + (size_t)t) * 64;

    // A fragments (16x32 f16 per chunk): lane holds K runs [k1,k1+8) and +16
    v16h af[4];
#pragma unroll
    for (int c = 0; c < 4; ++c) {
      int koff = (c & 1) * 32 + hi * 8;
#pragma unroll
      for (int run = 0; run < 2; ++run) {
        int k = koff + run * 16;
        float4 m0 = *(const float4*)(masks + vbase + k);
        float4 m1 = *(const float4*)(masks + vbase + k + 4);
        float e0,e1,e2,e3,e4,e5,e6,e7;
        if (c < 2) {
          float4 v0 = *(const float4*)(values + vbase + k);
          float4 v1 = *(const float4*)(values + vbase + k + 4);
          e0 = v0.x*m0.x; e1 = v0.y*m0.y; e2 = v0.z*m0.z; e3 = v0.w*m0.w;
          e4 = v1.x*m1.x; e5 = v1.y*m1.y; e6 = v1.z*m1.z; e7 = v1.w*m1.w;
        } else {
          e0 = m0.x; e1 = m0.y; e2 = m0.z; e3 = m0.w;
          e4 = m1.x; e5 = m1.y; e6 = m1.z; e7 = m1.w;
        }
        int j = run * 8;
        af[c][j+0]=(_Float16)e0; af[c][j+1]=(_Float16)e1;
        af[c][j+2]=(_Float16)e2; af[c][j+3]=(_Float16)e3;
        af[c][j+4]=(_Float16)e4; af[c][j+5]=(_Float16)e5;
        af[c][j+6]=(_Float16)e6; af[c][j+7]=(_Float16)e7;
      }
    }

    // time values for this lane's output rows (m = r + hi*8)
    float tv[8];
#pragma unroll
    for (int r = 0; r < 8; ++r)
      tv[r] = times[(size_t)(btile * 16 + r + hi * 8) * 1024 + t];

    size_t rowbase = (size_t)t * 256 + (size_t)btile * 16;
    for (int tile = 0; tile < 24; ++tile) {
      int g = tile * 16 + mlo;
      float wl = wlast[g], bi = bias[g];
      v8f acc;
#pragma unroll
      for (int r = 0; r < 8; ++r) acc[r] = tv[r] * wl + bi;  // time + bias seed
#pragma unroll
      for (int c = 0; c < 4; ++c) {
        const _Float16* wp = Wi + g * 128 + c * 32 + hi * 8;
        v16h bf = cat8(*(const v8h*)wp, *(const v8h*)(wp + 16));
        acc = WMMA_F16(af[c], bf, acc);
      }
#pragma unroll
      for (int r = 0; r < 8; ++r)
        gi[(rowbase + r + hi * 8) * 384 + g] = acc[r];
    }
  }
}

// ---------------------------------------------------------------------------
// Phase 2: GRU scan. grid = 16 WGs (one per 16-batch tile) x 8 waves.
// Wave w owns gate tiles {3w,3w+1,3w+2} of 24 (0-7:r, 8-15:z, 16-23:n).
// W_hh B-fragments resident in VGPRs; h in LDS (f32 + f16 mirror for A frags).
// gi is software-pipelined one step ahead, prefetched two ahead.
// ---------------------------------------------------------------------------
__global__ __launch_bounds__(256) void gru_scan_kernel(
    const float* __restrict__ gi, const int* __restrict__ lengths,
    const float* __restrict__ W_hh, const float* __restrict__ b_hh,
    const float* __restrict__ W_mu, const float* __restrict__ b_mu,
    const float* __restrict__ W_ls, const float* __restrict__ b_ls,
    float* __restrict__ out)
{
  __shared__ __align__(16) float    h_s[16 * 128];
  __shared__ __align__(16) _Float16 hf [16 * 128];
  __shared__ float rbuf[16 * 128];
  __shared__ float zbuf[16 * 128];
  __shared__ int   len_s[16];

  const int lane  = threadIdx.x & 31;
  const int wave  = threadIdx.x >> 5;
  const int hi    = lane >> 4;
  const int mlo   = lane & 15;
  const int btile = blockIdx.x;
  const int gt0   = wave * 3;

  for (int i = threadIdx.x; i < 16 * 128; i += 256) { h_s[i] = 0.f; hf[i] = (_Float16)0.f; }
  if (threadIdx.x < 16) len_s[threadIdx.x] = lengths[btile * 16 + threadIdx.x];

  // resident W_hh fragments: 12 x v16h = 96 VGPRs/lane
  v16h bw[12];
#pragma unroll
  for (int i = 0; i < 3; ++i) {
    int g = (gt0 + i) * 16 + mlo;
#pragma unroll
    for (int c = 0; c < 4; ++c) {
      const float* wp = W_hh + (size_t)g * 128 + c * 32 + hi * 16;
      float4 x0 = *(const float4*)(wp);
      float4 x1 = *(const float4*)(wp + 4);
      float4 x2 = *(const float4*)(wp + 8);
      float4 x3 = *(const float4*)(wp + 12);
      v16h f;
      f[0]=(_Float16)x0.x; f[1]=(_Float16)x0.y; f[2] =(_Float16)x0.z; f[3] =(_Float16)x0.w;
      f[4]=(_Float16)x1.x; f[5]=(_Float16)x1.y; f[6] =(_Float16)x1.z; f[7] =(_Float16)x1.w;
      f[8]=(_Float16)x2.x; f[9]=(_Float16)x2.y; f[10]=(_Float16)x2.z; f[11]=(_Float16)x2.w;
      f[12]=(_Float16)x3.x;f[13]=(_Float16)x3.y;f[14]=(_Float16)x3.z; f[15]=(_Float16)x3.w;
      bw[i * 4 + c] = f;
    }
  }
  float bhh[3];
#pragma unroll
  for (int i = 0; i < 3; ++i) bhh[i] = b_hh[(gt0 + i) * 16 + mlo];

  __syncthreads();

  float gcur[3][8], gnext[3][8];
  {
    size_t rowbase = (size_t)btile * 16;
#pragma unroll
    for (int i = 0; i < 3; ++i) {
      int g = (gt0 + i) * 16 + mlo;
#pragma unroll
      for (int r = 0; r < 8; ++r) gcur[i][r] = gi[(rowbase + r + hi * 8) * 384 + g];
    }
  }

  for (int t = 0; t < 1024; ++t) {
    __syncthreads();  // h stable; prev-step r/z bufs consumed

    // accumulator seed: r/z tiles get gi+b_hh; n tiles get b_hh only
    v8f acc[3];
#pragma unroll
    for (int i = 0; i < 3; ++i) {
      bool isn = ((gt0 + i) >> 3) == 2;
#pragma unroll
      for (int r = 0; r < 8; ++r) acc[i][r] = isn ? bhh[i] : (gcur[i][r] + bhh[i]);
    }
    // gh = h @ W_hh^T  (A fragments from f16 h in LDS)
#pragma unroll
    for (int c = 0; c < 4; ++c) {
      const _Float16* hp = hf + mlo * 128 + c * 32 + hi * 16;
      v16h a = cat8(*(const v8h*)hp, *(const v8h*)(hp + 8));
#pragma unroll
      for (int i = 0; i < 3; ++i) acc[i] = WMMA_F16(a, bw[i * 4 + c], acc[i]);
    }

    // pipeline: pull next step's gi while WMMAs drain
    if (t + 1 < 1024) {
      size_t rowbase = (size_t)(t + 1) * 256 + (size_t)btile * 16;
#pragma unroll
      for (int i = 0; i < 3; ++i) {
        int g = (gt0 + i) * 16 + mlo;
#pragma unroll
        for (int r = 0; r < 8; ++r) gnext[i][r] = gi[(rowbase + r + hi * 8) * 384 + g];
      }
      if (t + 2 < 1024) {
        const float* pf = gi + ((size_t)(t + 2) * 256 + btile * 16 + hi * 8) * 384
                             + (size_t)gt0 * 16 + mlo;
        __builtin_prefetch(pf, 0, 1);   // global_prefetch_b8
      }
    }

    // r/z producers
#pragma unroll
    for (int i = 0; i < 3; ++i) {
      int gt = gt0 + i, type = gt >> 3, col = (gt & 7) * 16 + mlo;
      if (type == 0) {
#pragma unroll
        for (int r = 0; r < 8; ++r)
          rbuf[(r + hi * 8) * 128 + col] = 1.f / (1.f + __expf(-acc[i][r]));
      } else if (type == 1) {
#pragma unroll
        for (int r = 0; r < 8; ++r)
          zbuf[(r + hi * 8) * 128 + col] = 1.f / (1.f + __expf(-acc[i][r]));
      }
    }
    __syncthreads();

    // n consumers + h update:  n = tanh(i_n + r*h_n);  h = (1-z)n + z h
#pragma unroll
    for (int i = 0; i < 3; ++i) {
      int gt = gt0 + i;
      if ((gt >> 3) == 2) {
        int col = (gt & 7) * 16 + mlo;
#pragma unroll
        for (int r = 0; r < 8; ++r) {
          int m = r + hi * 8;
          float rr = rbuf[m * 128 + col];
          float nn = tanhf(gcur[i][r] + rr * acc[i][r]);
          float zz = zbuf[m * 128 + col];
          float ho = h_s[m * 128 + col];
          float hn = (1.f - zz) * nn + zz * ho;
          hn = (t < len_s[m]) ? hn : ho;          // freeze past length
          h_s[m * 128 + col] = hn;
          hf [m * 128 + col] = (_Float16)hn;
        }
      }
    }
    if (t + 1 < 1024) {
#pragma unroll
      for (int i = 0; i < 3; ++i)
#pragma unroll
        for (int r = 0; r < 8; ++r) gcur[i][r] = gnext[i][r];
    }
  }
  __syncthreads();

  // output heads: mu = h W_mu^T + b_mu ; sigma = exp(clip(h W_ls^T + b_ls))
  if (threadIdx.x < 192) {
    int bl = threadIdx.x / 12;
    int zz = threadIdx.x - bl * 12;
    float mu = b_mu[zz], ls = b_ls[zz];
    for (int j = 0; j < 128; ++j) {
      float hv = h_s[bl * 128 + j];
      mu += hv * W_mu[zz * 128 + j];
      ls += hv * W_ls[zz * 128 + j];
    }
    int b = btile * 16 + bl;
    out[b * 12 + zz] = mu;                                  // mu [256,12]
    ls = fminf(fmaxf(ls, -5.f), 0.5f);
    out[256 * 12 + b * 12 + zz] = __expf(ls);               // sigma [256,12]
  }
}

// ---------------------------------------------------------------------------
extern "C" void kernel_launch(void* const* d_in, const int* in_sizes, int n_in,
                              void* d_out, int out_size, void* d_ws, size_t ws_size,
                              hipStream_t stream) {
  (void)in_sizes; (void)n_in; (void)out_size; (void)ws_size;
  const float* times   = (const float*)d_in[0];
  const float* values  = (const float*)d_in[1];
  const float* masks   = (const float*)d_in[2];
  const int*   lengths = (const int*)  d_in[3];
  const float* W_ih    = (const float*)d_in[4];
  const float* b_ih    = (const float*)d_in[5];
  const float* W_hh    = (const float*)d_in[6];
  const float* b_hh    = (const float*)d_in[7];
  const float* W_mu    = (const float*)d_in[8];
  const float* b_mu    = (const float*)d_in[9];
  const float* W_ls    = (const float*)d_in[10];
  const float* b_ls    = (const float*)d_in[11];
  float* out = (float*)d_out;
  float* gi  = (float*)d_ws;   // 256*1024*384 f32 = 384 MiB scratch, t-major

  size_t smem1 = (size_t)384 * 128 * 2 + (size_t)384 * 4 * 2;  // 101376 B
  gi_gemm_kernel<<<256, 256, smem1, stream>>>(times, values, masks, W_ih, b_ih, gi);
  gru_scan_kernel<<<16, 256, 0, stream>>>(gi, lengths, W_hh, b_hh,
                                          W_mu, b_mu, W_ls, b_ls, out);
}
